// QCConv_78151224918250
// MI455X (gfx1250) — matrix operand order repaired and checked by reading, not compiled, not run on hardware
//
#include <hip/hip_runtime.h>
#include <stdint.h>

#define N_NODES 50000
#define N_EDGES 200000
#define DIN 128
#define DOUT 128
#define NH 4
#define EPS_F 1e-5f
#define SLOPE 0.01f

typedef __bf16 bf16_t;
typedef bf16_t v16bf __attribute__((ext_vector_type(16)));
typedef float  v8f   __attribute__((ext_vector_type(8)));

__device__ __forceinline__ unsigned short f2bfb(float f) {
  union { float f; unsigned int u; } c; c.f = f;
  unsigned int u = c.u;
  u += 0x7fffu + ((u >> 16) & 1u);           // round-to-nearest-even
  return (unsigned short)(u >> 16);
}
__device__ __forceinline__ float bfb2f(unsigned short s) {
  union { unsigned int u; float f; } c; c.u = ((unsigned int)s) << 16; return c.f;
}
__device__ __forceinline__ v8f zero8() {
  v8f z;
#pragma unroll
  for (int i = 0; i < 8; ++i) z[i] = 0.f;
  return z;
}

// ---- WMMA fragment builders (CDNA5 wave32 layouts, CDNA5 ISA) ----
// A 16x32 bf16 per lane: K = kb+0..7 (elems 0-7) and kb+16..23 (elems 8-15),
// kb = 0 (lanes 0-15) or 8 (lanes 16-31)  -> two contiguous 16B runs.
__device__ __forceinline__ v16bf load_a(const unsigned short* p, int ld) {
  int lane = threadIdx.x & 31;
  int m = lane & 15;
  int kb = (lane & 16) ? 8 : 0;
  union { uint4 u[2]; v16bf v; } x;
  x.u[0] = *((const uint4*)(p + m * ld + kb));
  x.u[1] = *((const uint4*)(p + m * ld + kb + 16));
  return x.v;
}
// B 32x16 bf16 from a K-innermost (transposed) chunk sm[n][k]:
// lane n holds K = kb..kb+7 and kb+16..kb+23 (kb = 0 or 8) -> two 16B runs.
__device__ __forceinline__ v16bf load_bT(const unsigned short* pT, int ld) {
  int lane = threadIdx.x & 31;
  int n = lane & 15;
  int kb = (lane & 16) ? 8 : 0;
  union { uint4 u[2]; v16bf v; } x;
  x.u[0] = *((const uint4*)(pT + n * ld + kb));
  x.u[1] = *((const uint4*)(pT + n * ld + kb + 16));
  return x.v;
}
__device__ __forceinline__ v8f wmma_bf16(v16bf a, v16bf b, v8f c) {
  return __builtin_amdgcn_wmma_f32_16x16x32_bf16(false, a, false, b, (short)0, c,
                                                 false, false);
}

// ---- weights f32 [B][K][N] -> bf16 transposed [B][N][K] ----
__global__ void cvt_t_bf16(const float* __restrict__ s, unsigned short* __restrict__ d,
                           int K, int N, int total) {
  int i = blockIdx.x * blockDim.x + threadIdx.x;
  if (i >= total) return;
  int kn = K * N;
  int b = i / kn;
  int rem = i - b * kn;
  int n = rem / K;
  int k = rem - n * K;
  d[i] = f2bfb(s[((size_t)b * K + k) * N + n]);
}

// ---- node projections: Kv/Vv[h] = x @ W[h]  (grid.y = head*2+proj) ----
__global__ __launch_bounds__(128) void node_proj_kernel(
    const float* __restrict__ x, const unsigned short* __restrict__ wKT,
    const unsigned short* __restrict__ wVT,
    unsigned short* __restrict__ Kv, unsigned short* __restrict__ Vv) {
  __shared__ __align__(16) unsigned short sm_a[16][136];
  __shared__ __align__(16) unsigned short sm_cT[128][40];
  const int tid = threadIdx.x, lane = tid & 31, wave = tid >> 5;
  const int row0 = blockIdx.x * 16;
  const int head = blockIdx.y >> 1, proj = blockIdx.y & 1;
  const unsigned short* W = (proj ? wVT : wKT) + (size_t)head * DIN * DOUT;  // [n][k]
  unsigned short* out = (proj ? Vv : Kv) + (size_t)head * N_NODES * DOUT;

  for (int it = tid; it < 512; it += 128) {
    int r = it >> 5, c4 = it & 31;
    float4 v = ((const float4*)(x + (size_t)(row0 + r) * DIN))[c4];
    int c = c4 * 4;
    sm_a[r][c] = f2bfb(v.x); sm_a[r][c + 1] = f2bfb(v.y);
    sm_a[r][c + 2] = f2bfb(v.z); sm_a[r][c + 3] = f2bfb(v.w);
  }
  v8f a0 = zero8(), a1 = zero8();
  for (int kk = 0; kk < 4; ++kk) {
    __syncthreads();
    for (int it = tid; it < 512; it += 128) {
      int n = it >> 2, q = it & 3;
      *((uint4*)&sm_cT[n][q * 8]) =
          ((const uint4*)(W + (size_t)n * DIN + kk * 32))[q];
    }
    __syncthreads();
    v16bf a = load_a(&sm_a[0][kk * 32], 136);
    a0 = wmma_bf16(a, load_bT(&sm_cT[wave * 16][0], 40), a0);
    a1 = wmma_bf16(a, load_bT(&sm_cT[(wave + 4) * 16][0], 40), a1);
  }
  int n = lane & 15, mb = (lane & 16) ? 8 : 0;
#pragma unroll
  for (int r = 0; r < 8; ++r) {
    int m = row0 + mb + r;
    out[(size_t)m * DOUT + wave * 16 + n] = f2bfb(a0[r]);
    out[(size_t)m * DOUT + (wave + 4) * 16 + n] = f2bfb(a1[r]);
  }
}

// ---- fused per-edge kernel: 16 edges x 1 head per block, 8 waves ----
__global__ __launch_bounds__(256) void edge_kernel(
    const int* __restrict__ eidx, const float* __restrict__ ef,
    const unsigned short* __restrict__ Kv, const unsigned short* __restrict__ Vv,
    const unsigned short* __restrict__ wKeT, const unsigned short* __restrict__ wVeT,
    const unsigned short* __restrict__ wLuT, const float* __restrict__ luB,
    const float* __restrict__ g1, const float* __restrict__ b1,
    const unsigned short* __restrict__ wMsgT, const float* __restrict__ msgB,
    const float* __restrict__ g2, const float* __restrict__ b2,
    float* __restrict__ hout) {
  __shared__ __align__(16) unsigned short sm_ef[16][136];
  __shared__ __align__(16) unsigned short sm_val[16][264];
  __shared__ __align__(16) unsigned short sm_t[16][264];
  __shared__ __align__(16) float sm_g[16][256];            // alpha -> gate -> m
  __shared__ __align__(16) unsigned short sm_q[16][136];   // bf16 q
  __shared__ __align__(16) unsigned short sm_cT[256][40];  // transposed weight chunk
  __shared__ int sm_src[16], sm_dst[16];

  const int tid = threadIdx.x, lane = tid & 31, wave = tid >> 5;
  const int h = blockIdx.y;
  const int e0 = blockIdx.x * 16;
  const float scale = 0.0625f;  // 1/sqrt(256)

  if (tid < 16) { sm_src[tid] = eidx[e0 + tid]; sm_dst[tid] = eidx[N_EDGES + e0 + tid]; }
  __syncthreads();

  {  // gather q = Kv[dst] and val[:,0:128] = Vv[src] (bf16 uint4 copies)
    int r = tid >> 4, g8 = tid & 15;
    *((uint4*)&sm_q[r][g8 * 8]) =
        ((const uint4*)(Kv + ((size_t)h * N_NODES + sm_dst[r]) * DOUT))[g8];
    *((uint4*)&sm_val[r][g8 * 8]) =
        ((const uint4*)(Vv + ((size_t)h * N_NODES + sm_src[r]) * DOUT))[g8];
  }
  for (int it = tid; it < 512; it += 256) {  // edge_feature tile -> bf16
    int r = it >> 5, c4 = it & 31;
    float4 v = ((const float4*)(ef + (size_t)(e0 + r) * DIN))[c4];
    int c = c4 * 4;
    sm_ef[r][c] = f2bfb(v.x); sm_ef[r][c + 1] = f2bfb(v.y);
    sm_ef[r][c + 2] = f2bfb(v.z); sm_ef[r][c + 3] = f2bfb(v.w);
  }
  __syncthreads();

  {  // alpha low half = q * Kv[src] * scale
    int r = tid >> 4, g8 = tid & 15;
    uint4 uk = ((const uint4*)(Kv + ((size_t)h * N_NODES + sm_src[r]) * DOUT))[g8];
    const unsigned short* us = (const unsigned short*)&uk;
#pragma unroll
    for (int i = 0; i < 8; ++i) {
      int c = g8 * 8 + i;
      sm_g[r][c] = bfb2f(sm_q[r][c]) * bfb2f(us[i]) * scale;
    }
  }

  // KE = ef @ Ke2v[h]; wave owns col tile `wave`; then alpha high half
  v8f acc = zero8();
  for (int kk = 0; kk < 4; ++kk) {
    __syncthreads();
    for (int it = tid; it < 512; it += 256) {
      int n = it >> 2, q = it & 3;
      *((uint4*)&sm_cT[n][q * 8]) =
          ((const uint4*)(wKeT + ((size_t)h * DOUT + n) * DIN + kk * 32))[q];
    }
    __syncthreads();
    acc = wmma_bf16(load_a(&sm_ef[0][kk * 32], 136), load_bT(&sm_cT[wave * 16][0], 40), acc);
  }
  {
    int n = lane & 15, mb = (lane & 16) ? 8 : 0;
    int col = wave * 16 + n;
#pragma unroll
    for (int r = 0; r < 8; ++r) {
      int m = mb + r;
      sm_g[m][128 + col] = bfb2f(sm_q[m][col]) * acc[r] * scale;
    }
  }
  // VE = ef @ Ve2v[h] -> val[:,128:256] (bf16)
  acc = zero8();
  for (int kk = 0; kk < 4; ++kk) {
    __syncthreads();
    for (int it = tid; it < 512; it += 256) {
      int n = it >> 2, q = it & 3;
      *((uint4*)&sm_cT[n][q * 8]) =
          ((const uint4*)(wVeT + ((size_t)h * DOUT + n) * DIN + kk * 32))[q];
    }
    __syncthreads();
    acc = wmma_bf16(load_a(&sm_ef[0][kk * 32], 136), load_bT(&sm_cT[wave * 16][0], 40), acc);
  }
  {
    int n = lane & 15, mb = (lane & 16) ? 8 : 0;
    int col = wave * 16 + n;
#pragma unroll
    for (int r = 0; r < 8; ++r) sm_val[mb + r][128 + col] = f2bfb(acc[r]);
  }
  __syncthreads();

  // LN1 over 256 + sigmoid -> gate (wave per 2 rows, lane-parallel)
#pragma unroll
  for (int rr = 0; rr < 2; ++rr) {
    int r = wave * 2 + rr;
    float v[8], s = 0.f, s2 = 0.f;
#pragma unroll
    for (int i = 0; i < 8; ++i) { float t = sm_g[r][lane * 8 + i]; v[i] = t; s += t; s2 += t * t; }
#pragma unroll
    for (int off = 16; off > 0; off >>= 1) { s += __shfl_xor(s, off, 32); s2 += __shfl_xor(s2, off, 32); }
    float mean = s * (1.f / 256.f);
    float inv = rsqrtf(s2 * (1.f / 256.f) - mean * mean + EPS_F);
#pragma unroll
    for (int i = 0; i < 8; ++i) {
      int c = lane * 8 + i;
      float z = (v[i] - mean) * inv * g1[h * 256 + c] + b1[h * 256 + c];
      sm_g[r][c] = 1.0f / (1.0f + __expf(-z));
    }
  }

  // t = (val @ lu_W[h] + lu_b) * gate ; wave owns col tiles {wave, wave+8}
  v8f a0 = zero8(), a1 = zero8();
  for (int kk = 0; kk < 8; ++kk) {
    __syncthreads();
    for (int it = tid; it < 1024; it += 256) {
      int n = it >> 2, q = it & 3;
      *((uint4*)&sm_cT[n][q * 8]) =
          ((const uint4*)(wLuT + ((size_t)h * 256 + n) * 256 + kk * 32))[q];
    }
    if (kk < 7)  // warm next chunk (global_prefetch_b8)
      __builtin_prefetch(wLuT + ((size_t)h * 256 + tid) * 256 + (kk + 1) * 32, 0, 1);
    __syncthreads();
    v16bf a = load_a(&sm_val[0][kk * 32], 264);
    a0 = wmma_bf16(a, load_bT(&sm_cT[wave * 16][0], 40), a0);
    a1 = wmma_bf16(a, load_bT(&sm_cT[(wave + 8) * 16][0], 40), a1);
  }
  {
    int n = lane & 15, mb = (lane & 16) ? 8 : 0;
    int c0 = wave * 16 + n, c1 = (wave + 8) * 16 + n;
    float bi0 = luB[h * 256 + c0], bi1 = luB[h * 256 + c1];
#pragma unroll
    for (int r = 0; r < 8; ++r) {
      int m = mb + r;
      sm_t[m][c0] = f2bfb((a0[r] + bi0) * sm_g[m][c0]);
      sm_t[m][c1] = f2bfb((a1[r] + bi1) * sm_g[m][c1]);
    }
  }

  // m = t @ msg_W[h] + msg_b ; wave owns col tile `wave`
  v8f mac = zero8();
  for (int kk = 0; kk < 8; ++kk) {
    __syncthreads();
    for (int it = tid; it < 512; it += 256) {
      int n = it >> 2, q = it & 3;
      *((uint4*)&sm_cT[n][q * 8]) =
          ((const uint4*)(wMsgT + ((size_t)h * DOUT + n) * 256 + kk * 32))[q];
    }
    if (kk < 7 && tid < 128)
      __builtin_prefetch(wMsgT + ((size_t)h * DOUT + tid) * 256 + (kk + 1) * 32, 0, 1);
    __syncthreads();
    mac = wmma_bf16(load_a(&sm_t[0][kk * 32], 264), load_bT(&sm_cT[wave * 16][0], 40), mac);
  }
  {
    int n = lane & 15, mb = (lane & 16) ? 8 : 0;
    int col = wave * 16 + n;
    float bias = msgB[h * 128 + col];
#pragma unroll
    for (int r = 0; r < 8; ++r) sm_g[mb + r][col] = mac[r] + bias;
  }
  __syncthreads();

  // LN2 over 128 + affine + leaky + segment-sum scatter
#pragma unroll
  for (int rr = 0; rr < 2; ++rr) {
    int r = wave * 2 + rr;
    float v[4], s = 0.f, s2 = 0.f;
#pragma unroll
    for (int i = 0; i < 4; ++i) { float t = sm_g[r][lane * 4 + i]; v[i] = t; s += t; s2 += t * t; }
#pragma unroll
    for (int off = 16; off > 0; off >>= 1) { s += __shfl_xor(s, off, 32); s2 += __shfl_xor(s2, off, 32); }
    float mean = s * (1.f / 128.f);
    float inv = rsqrtf(s2 * (1.f / 128.f) - mean * mean + EPS_F);
    float* orow = hout + (size_t)sm_dst[r] * (DOUT * NH) + h * DOUT;
#pragma unroll
    for (int i = 0; i < 4; ++i) {
      int c = lane * 4 + i;
      float z = (v[i] - mean) * inv * g2[h * 128 + c] + b2[h * 128 + c];
      z = z > 0.f ? z : SLOPE * z;
      atomicAdd(&orow[c], z);
    }
  }
}

// ---- cat projection + BN statistics ----
__global__ __launch_bounds__(128) void cat_kernel(
    const float* __restrict__ hv, const unsigned short* __restrict__ wCatT,
    const float* __restrict__ catB, float* __restrict__ out_pre,
    float* __restrict__ bn_sum, float* __restrict__ bn_sq) {
  __shared__ __align__(16) unsigned short sm_a[16][520];
  __shared__ __align__(16) unsigned short sm_cT[128][40];
  const int tid = threadIdx.x, lane = tid & 31, wave = tid >> 5;
  const int row0 = blockIdx.x * 16;
  for (int it = tid; it < 2048; it += 128) {
    int r = it >> 7, c4 = it & 127;
    float4 v = ((const float4*)(hv + (size_t)(row0 + r) * 512))[c4];
    int c = c4 * 4;
    sm_a[r][c] = f2bfb(v.x); sm_a[r][c + 1] = f2bfb(v.y);
    sm_a[r][c + 2] = f2bfb(v.z); sm_a[r][c + 3] = f2bfb(v.w);
  }
  v8f a0 = zero8(), a1 = zero8();
  for (int kk = 0; kk < 16; ++kk) {
    __syncthreads();
    for (int it = tid; it < 512; it += 128) {
      int n = it >> 2, q = it & 3;
      *((uint4*)&sm_cT[n][q * 8]) =
          ((const uint4*)(wCatT + (size_t)n * 512 + kk * 32))[q];
    }
    __syncthreads();
    v16bf a = load_a(&sm_a[0][kk * 32], 520);
    a0 = wmma_bf16(a, load_bT(&sm_cT[wave * 16][0], 40), a0);
    a1 = wmma_bf16(a, load_bT(&sm_cT[(wave + 4) * 16][0], 40), a1);
  }
  int n = lane & 15, mb = (lane & 16) ? 8 : 0;
  int c0 = wave * 16 + n, c1 = (wave + 4) * 16 + n;
  float bi0 = catB[c0], bi1 = catB[c1];
  float s0 = 0.f, q0 = 0.f, s1 = 0.f, q1 = 0.f;
#pragma unroll
  for (int r = 0; r < 8; ++r) {
    int m = row0 + mb + r;
    float v0 = a0[r] + bi0, v1 = a1[r] + bi1;
    out_pre[(size_t)m * 128 + c0] = v0;
    out_pre[(size_t)m * 128 + c1] = v1;
    s0 += v0; q0 += v0 * v0; s1 += v1; q1 += v1 * v1;
  }
  s0 += __shfl_xor(s0, 16, 32); q0 += __shfl_xor(q0, 16, 32);
  s1 += __shfl_xor(s1, 16, 32); q1 += __shfl_xor(q1, 16, 32);
  if (lane < 16) {
    atomicAdd(&bn_sum[c0], s0); atomicAdd(&bn_sq[c0], q0);
    atomicAdd(&bn_sum[c1], s1); atomicAdd(&bn_sq[c1], q1);
  }
}

// ---- BN + leaky + residual (x @ inp_W + b) ----
__global__ __launch_bounds__(128) void final_kernel(
    const float* __restrict__ x, const unsigned short* __restrict__ wInpT,
    const float* __restrict__ inpB, const float* __restrict__ out_pre,
    const float* __restrict__ bn_sum, const float* __restrict__ bn_sq,
    const float* __restrict__ bnG, const float* __restrict__ bnB,
    float* __restrict__ out) {
  __shared__ __align__(16) unsigned short sm_a[16][136];
  __shared__ __align__(16) unsigned short sm_cT[128][40];
  const int tid = threadIdx.x, lane = tid & 31, wave = tid >> 5;
  const int row0 = blockIdx.x * 16;
  for (int it = tid; it < 512; it += 128) {
    int r = it >> 5, c4 = it & 31;
    float4 v = ((const float4*)(x + (size_t)(row0 + r) * 128))[c4];
    int c = c4 * 4;
    sm_a[r][c] = f2bfb(v.x); sm_a[r][c + 1] = f2bfb(v.y);
    sm_a[r][c + 2] = f2bfb(v.z); sm_a[r][c + 3] = f2bfb(v.w);
  }
  v8f a0 = zero8(), a1 = zero8();
  for (int kk = 0; kk < 4; ++kk) {
    __syncthreads();
    for (int it = tid; it < 512; it += 128) {
      int n = it >> 2, q = it & 3;
      *((uint4*)&sm_cT[n][q * 8]) =
          ((const uint4*)(wInpT + (size_t)n * 128 + kk * 32))[q];
    }
    __syncthreads();
    v16bf a = load_a(&sm_a[0][kk * 32], 136);
    a0 = wmma_bf16(a, load_bT(&sm_cT[wave * 16][0], 40), a0);
    a1 = wmma_bf16(a, load_bT(&sm_cT[(wave + 4) * 16][0], 40), a1);
  }
  int n = lane & 15, mb = (lane & 16) ? 8 : 0;
  int c0 = wave * 16 + n, c1 = (wave + 4) * 16 + n;
  float ib0 = inpB[c0], ib1 = inpB[c1];
  float m0 = bn_sum[c0] * (1.f / N_NODES), m1 = bn_sum[c1] * (1.f / N_NODES);
  float iv0 = rsqrtf(bn_sq[c0] * (1.f / N_NODES) - m0 * m0 + EPS_F);
  float iv1 = rsqrtf(bn_sq[c1] * (1.f / N_NODES) - m1 * m1 + EPS_F);
  float gg0 = bnG[c0], gg1 = bnG[c1], bb0 = bnB[c0], bb1 = bnB[c1];
#pragma unroll
  for (int r = 0; r < 8; ++r) {
    int m = row0 + mb + r;
    float z0 = (out_pre[(size_t)m * 128 + c0] - m0) * iv0 * gg0 + bb0;
    float z1 = (out_pre[(size_t)m * 128 + c1] - m1) * iv1 * gg1 + bb1;
    z0 = z0 > 0.f ? z0 : SLOPE * z0;
    z1 = z1 > 0.f ? z1 : SLOPE * z1;
    out[(size_t)m * 128 + c0] = z0 + a0[r] + ib0;
    out[(size_t)m * 128 + c1] = z1 + a1[r] + ib1;
  }
}

extern "C" void kernel_launch(void* const* d_in, const int* in_sizes, int n_in,
                              void* d_out, int out_size, void* d_ws, size_t ws_size,
                              hipStream_t stream) {
  (void)in_sizes; (void)n_in; (void)out_size; (void)ws_size;
  const float* x    = (const float*)d_in[0];
  const int*   eidx = (const int*)d_in[1];
  const float* ef   = (const float*)d_in[2];
  const float* Kv2v = (const float*)d_in[3];
  const float* Ke2v = (const float*)d_in[4];
  const float* Vv2v = (const float*)d_in[5];
  const float* Ve2v = (const float*)d_in[6];
  const float* luW  = (const float*)d_in[7];
  const float* luB  = (const float*)d_in[8];
  const float* g1   = (const float*)d_in[9];
  const float* b1   = (const float*)d_in[10];
  const float* msgW = (const float*)d_in[11];
  const float* msgB = (const float*)d_in[12];
  const float* g2   = (const float*)d_in[13];
  const float* b2   = (const float*)d_in[14];
  const float* catW = (const float*)d_in[15];
  const float* catB = (const float*)d_in[16];
  const float* inpW = (const float*)d_in[17];
  const float* inpB = (const float*)d_in[18];
  const float* bnG  = (const float*)d_in[19];
  const float* bnB  = (const float*)d_in[20];

  char* ws = (char*)d_ws;
  size_t off = 0;
  auto alloc = [&](size_t bytes) {
    char* p = ws + off;
    off = (off + bytes + 255) & ~(size_t)255;
    return p;
  };
  unsigned short* wKvT  = (unsigned short*)alloc((size_t)NH * 128 * 128 * 2);
  unsigned short* wKeT  = (unsigned short*)alloc((size_t)NH * 128 * 128 * 2);
  unsigned short* wVvT  = (unsigned short*)alloc((size_t)NH * 128 * 128 * 2);
  unsigned short* wVeT  = (unsigned short*)alloc((size_t)NH * 128 * 128 * 2);
  unsigned short* wLuT  = (unsigned short*)alloc((size_t)NH * 256 * 256 * 2);
  unsigned short* wMsgT = (unsigned short*)alloc((size_t)NH * 128 * 256 * 2);
  unsigned short* wCatT = (unsigned short*)alloc((size_t)128 * 512 * 2);
  unsigned short* wInpT = (unsigned short*)alloc((size_t)128 * 128 * 2);
  unsigned short* KvN   = (unsigned short*)alloc((size_t)NH * N_NODES * 128 * 2);
  unsigned short* VvN   = (unsigned short*)alloc((size_t)NH * N_NODES * 128 * 2);
  float* hout = (float*)alloc((size_t)N_NODES * 512 * 4);
  float* opre = (float*)alloc((size_t)N_NODES * 128 * 4);
  float* bn   = (float*)alloc(256 * 4);

  hipMemsetAsync(hout, 0, (size_t)N_NODES * 512 * 4, stream);
  hipMemsetAsync(bn, 0, 256 * 4, stream);

  auto cvt = [&](const float* s, unsigned short* d, int K, int N, int total) {
    cvt_t_bf16<<<(total + 255) / 256, 256, 0, stream>>>(s, d, K, N, total);
  };
  cvt(Kv2v, wKvT, 128, 128, NH * 128 * 128);
  cvt(Ke2v, wKeT, 128, 128, NH * 128 * 128);
  cvt(Vv2v, wVvT, 128, 128, NH * 128 * 128);
  cvt(Ve2v, wVeT, 128, 128, NH * 128 * 128);
  cvt(luW,  wLuT, 256, 256, NH * 256 * 256);
  cvt(msgW, wMsgT, 256, 128, NH * 256 * 128);
  cvt(catW, wCatT, 512, 128, 512 * 128);
  cvt(inpW, wInpT, 128, 128, 128 * 128);

  node_proj_kernel<<<dim3(N_NODES / 16, 8), 128, 0, stream>>>(x, wKvT, wVvT, KvN, VvN);
  edge_kernel<<<dim3(N_EDGES / 16, NH), 256, 0, stream>>>(
      eidx, ef, KvN, VvN, wKeT, wVeT, wLuT, luB, g1, b1, wMsgT, msgB, g2, b2, hout);
  cat_kernel<<<N_NODES / 16, 128, 0, stream>>>(hout, wCatT, catB, opre, bn, bn + 128);
  final_kernel<<<N_NODES / 16, 128, 0, stream>>>(x, wInpT, inpB, opre, bn, bn + 128,
                                                 bnG, bnB, (float*)d_out);
}
